// NGramWindowAttention3D_9964324126881
// MI455X (gfx1250) — matrix operand order repaired and checked by reading, not compiled, not run on hardware
//
#include <hip/hip_runtime.h>
#include <hip/hip_bf16.h>

#define DIM   192
#define HEADS 6
#define NTOK  256
#define HD    32
#define BATCH 256
#define NWIN  64

typedef __attribute__((ext_vector_type(16))) __bf16 v16bf;
typedef __attribute__((ext_vector_type(8)))  float  v8f;

union BFrag { v16bf v; unsigned int u[8]; };

__device__ __forceinline__ unsigned int f2bf1(float f) {
  unsigned int u = __float_as_uint(f);
  return (u + 0x7FFFu + ((u >> 16) & 1u)) >> 16;   // RNE
}
__device__ __forceinline__ unsigned int pack2bf(float lo, float hi) {
  return f2bf1(lo) | (f2bf1(hi) << 16);
}

// Async global->LDS copy, 16B granularity (gfx1250 GLOBAL_LOAD_ASYNC_TO_LDS_B128,
// tracked by ASYNCcnt). lds_off = byte offset (low 32 bits of generic LDS addr).
__device__ __forceinline__ void async_copy_b128(unsigned int lds_off, const void* gaddr) {
  asm volatile("global_load_async_to_lds_b128 %0, %1, off"
               :: "v"(lds_off), "v"(gaddr) : "memory");
}
__device__ __forceinline__ void wait_async0() {
  asm volatile("s_wait_asynccnt 0" ::: "memory");
}

// A fragment (16 rows x 32 K, bf16 row-major). Lane L<16: row L, K 0-7 & 16-23.
// Lane L>=16: row L-16, K 8-15 & 24-31.  (ISA 7.12.2, 16-bit A 16x32)
__device__ __forceinline__ v16bf load_a_frag(const unsigned short* tile, int ldm) {
  int lane = threadIdx.x & 31;
  const unsigned short* p = tile + (lane & 15) * ldm + (lane >> 4) * 8;
  BFrag f;
  uint4 a = *(const uint4*)(p);
  uint4 b = *(const uint4*)(p + 16);
  f.u[0]=a.x; f.u[1]=a.y; f.u[2]=a.z; f.u[3]=a.w;
  f.u[4]=b.x; f.u[5]=b.y; f.u[6]=b.z; f.u[7]=b.w;
  return f.v;
}

// B fragment from transposed B (16 n-rows x 32 K, bf16 row-major).
// Lane L<16: column n=L, K 0-7 & 16-23; lane L>=16: column n=L-16, K 8-15 & 24-31.
__device__ __forceinline__ v16bf load_b_frag(const unsigned short* tileT, int ldm) {
  int lane = threadIdx.x & 31;
  const unsigned short* p = tileT + (lane & 15) * ldm + (lane >> 4) * 8;
  BFrag f;
  uint4 a = *(const uint4*)(p);
  uint4 b = *(const uint4*)(p + 16);
  f.u[0]=a.x; f.u[1]=a.y; f.u[2]=a.z; f.u[3]=a.w;
  f.u[4]=b.x; f.u[5]=b.y; f.u[6]=b.z; f.u[7]=b.w;
  return f.v;
}

#define WMMA_BF16(A,B,C) __builtin_amdgcn_wmma_f32_16x16x32_bf16(false,(A),false,(B),(short)0,(C),false,false)

// ---------------- prep: bias gather + weight transposes to bf16 ----------------
__global__ __launch_bounds__(256) void prep_kernel(
    const float* __restrict__ qkv_w, const float* __restrict__ proj_w,
    const float* __restrict__ bias_table, const int* __restrict__ rel_index,
    float* __restrict__ biasM, unsigned short* __restrict__ Wqkvt,
    unsigned short* __restrict__ Wpt)
{
  int i = blockIdx.x * blockDim.x + threadIdx.x;
  if (i < HEADS * NTOK * NTOK) {
    int h  = i / (NTOK * NTOK);
    int mn = i - h * (NTOK * NTOK);
    biasM[i] = bias_table[rel_index[mn] * HEADS + h];
  }
  if (i < DIM * 3 * DIM) {                 // qkv_w (192 x 576) -> Wqkvt (576 x 192) bf16
    int k = i / (3 * DIM), n = i % (3 * DIM);
    Wqkvt[n * DIM + k] = (unsigned short)f2bf1(qkv_w[i]);
  }
  if (i < DIM * DIM) {                     // proj_w (192 x 192) -> Wpt (192 x 192) bf16
    int k = i / DIM, n = i % DIM;
    Wpt[n * DIM + k] = (unsigned short)f2bf1(proj_w[i]);
  }
}

// ---------------- QKV projection: per batch, X(256x192) @ W -> Q,K,Vt bf16 ------
__global__ __launch_bounds__(128) void qkv_kernel(
    const float* __restrict__ x, const unsigned short* __restrict__ Wt,
    const float* __restrict__ qkv_b,
    unsigned short* __restrict__ Q, unsigned short* __restrict__ K,
    unsigned short* __restrict__ Vt)
{
  int b      = blockIdx.x >> 2;
  int chunk  = blockIdx.x & 3;
  int wave   = threadIdx.x >> 5;
  int lane   = threadIdx.x & 31;
  int half   = lane >> 4;
  int rowbase = chunk * 64 + wave * 16;
  int row    = rowbase + (lane & 15);
  const float* xrow = x + ((size_t)b * NTOK + row) * DIM;

  // preload A fragments for all 6 K-steps (fp32 -> bf16 on the fly)
  v16bf afr[6];
#pragma unroll
  for (int ks = 0; ks < 6; ++ks) {
    const float* p = xrow + ks * 32 + half * 8;
    float4 a0 = *(const float4*)(p);
    float4 a1 = *(const float4*)(p + 4);
    float4 b0 = *(const float4*)(p + 16);
    float4 b1 = *(const float4*)(p + 20);
    BFrag f;
    f.u[0]=pack2bf(a0.x,a0.y); f.u[1]=pack2bf(a0.z,a0.w);
    f.u[2]=pack2bf(a1.x,a1.y); f.u[3]=pack2bf(a1.z,a1.w);
    f.u[4]=pack2bf(b0.x,b0.y); f.u[5]=pack2bf(b0.z,b0.w);
    f.u[6]=pack2bf(b1.x,b1.y); f.u[7]=pack2bf(b1.z,b1.w);
    afr[ks] = f.v;
  }

  int coln = lane & 15;
  for (int nt = 0; nt < 36; ++nt) {
    __builtin_prefetch(Wt + (nt + 1) * 16 * DIM, 0, 3);   // global_prefetch_b8
    v8f acc = {};
#pragma unroll
    for (int ks = 0; ks < 6; ++ks) {
      v16bf bfr = load_b_frag(Wt + nt * 16 * DIM + ks * 32, DIM);
      acc = WMMA_BF16(afr[ks], bfr, acc);
    }
    // which / h are uniform per 16-wide n-tile (16 | 32 | 192 alignment)
    int which = nt / 12;                       // 0=q 1=k 2=v (scalar)
    int sub   = nt - which * 12;
    int h     = sub >> 1;                      // scalar
    int d     = coln + ((nt & 1) << 4);
    float bia = qkv_b[nt * 16 + coln];
    size_t bh = (size_t)b * HEADS + h;
    if (which == 0) {
#pragma unroll
      for (int r = 0; r < 8; ++r) {
        int token = rowbase + r + half * 8;
        float v = (acc[r] + bia) * 0.17677669529663687f;   // 1/sqrt(32) folded
        Q[(bh * NTOK + token) * HD + d] = (unsigned short)f2bf1(v);
      }
    } else if (which == 1) {
#pragma unroll
      for (int r = 0; r < 8; ++r) {
        int token = rowbase + r + half * 8;
        K[(bh * NTOK + token) * HD + d] = (unsigned short)f2bf1(acc[r] + bia);
      }
    } else {
#pragma unroll
      for (int r = 0; r < 8; ++r) {
        int token = rowbase + r + half * 8;
        Vt[(bh * HD + d) * NTOK + token] = (unsigned short)f2bf1(acc[r] + bia);
      }
    }
  }
}

// ---------------- attention: one workgroup per (b,h), 16 waves x 16 rows -------
__global__ __launch_bounds__(512) void attn_kernel(
    const unsigned short* __restrict__ Q, const unsigned short* __restrict__ K,
    const unsigned short* __restrict__ Vt, const float* __restrict__ biasM,
    const float* __restrict__ mask, unsigned short* __restrict__ AO)
{
  __shared__ unsigned short ldsK[NTOK * HD];   // 16KB: K  (256 tok x 32 d)
  __shared__ unsigned short ldsV[HD * NTOK];   // 16KB: V^T (32 d x 256 tok)
  __shared__ unsigned short pbuf[16 * 512];    // 16KB: per-wave P staging

  int bh   = blockIdx.x;
  int b    = bh / HEADS;
  int h    = bh - b * HEADS;
  int w    = b & (NWIN - 1);
  int tid  = threadIdx.x;
  int wave = tid >> 5;
  int lane = tid & 31;
  int half = lane >> 4;
  int coln = lane & 15;
  int m0   = wave * 16;

  const unsigned short* Qb = Q  + ((size_t)bh * NTOK + m0) * HD;
  const unsigned short* Kb = K  + (size_t)bh * NTOK * HD;
  const unsigned short* Vb = Vt + (size_t)bh * HD * NTOK;
  const float* biash = biasM + (size_t)h * NTOK * NTOK;
  const float* maskw = mask  + (size_t)w * NTOK * NTOK;

  // Stage K and V^T once per workgroup via async-to-LDS (ASYNCcnt-tracked):
  // 512 threads x 2 x b128 covers each 16KB region.
  {
    unsigned int lk = (unsigned int)(size_t)ldsK;   // LDS offset = addr[31:0]
    unsigned int lv = (unsigned int)(size_t)ldsV;
    const char* gk = (const char*)Kb;
    const char* gv = (const char*)Vb;
#pragma unroll
    for (int c = 0; c < 2; ++c) {
      unsigned int off = (unsigned int)(tid + c * 512) * 16;
      async_copy_b128(lk + off, gk + off);
      async_copy_b128(lv + off, gv + off);
    }
    wait_async0();
    __syncthreads();
  }

  v16bf qa = load_a_frag(Qb, HD);

  // S = Q @ K^T : 16 tiles of 16x16, hd=32 -> single K-step each (from LDS)
  v8f s[16];
#pragma unroll
  for (int t = 0; t < 16; ++t) {
    v16bf kb = load_b_frag(ldsK + t * 16 * HD, HD);
    v8f z = {};
    s[t] = WMMA_BF16(qa, kb, z);
  }

  // + bias + mask, row max (rows r in lanes 0-15, rows r+8 in lanes 16-31)
  float rmax[8], rinv[8];
#pragma unroll
  for (int r = 0; r < 8; ++r) {
    int m    = m0 + r + half * 8;
    int base = m * NTOK;
    float mx = -3.4e38f;
#pragma unroll
    for (int t = 0; t < 16; ++t) {
      int n = t * 16 + coln;
      float v = s[t][r] + biash[base + n] + maskw[base + n];
      s[t][r] = v;
      mx = fmaxf(mx, v);
    }
#pragma unroll
    for (int off = 8; off >= 1; off >>= 1)
      mx = fmaxf(mx, __shfl_xor(mx, off, 32));
    rmax[r] = mx;
  }
#pragma unroll
  for (int r = 0; r < 8; ++r) {
    float sum = 0.f;
#pragma unroll
    for (int t = 0; t < 16; ++t) {
      float e = __expf(s[t][r] - rmax[r]);
      s[t][r] = e;
      sum += e;
    }
#pragma unroll
    for (int off = 8; off >= 1; off >>= 1)
      sum += __shfl_xor(sum, off, 32);
    rinv[r] = 1.0f / sum;
  }

  // O = P @ V : round-trip P tiles through per-wave LDS to get A-layout bf16
  unsigned short* pw = pbuf + wave * 512;
  v8f o0 = {}, o1 = {};
#pragma unroll
  for (int sstep = 0; sstep < 8; ++sstep) {
#pragma unroll
    for (int j = 0; j < 2; ++j) {
      int t = sstep * 2 + j;
#pragma unroll
      for (int r = 0; r < 8; ++r)
        pw[(r + half * 8) * 32 + j * 16 + coln] =
            (unsigned short)f2bf1(s[t][r] * rinv[r]);
    }
    v16bf pa  = load_a_frag(pw, 32);                       // same-wave DS in-order
    v16bf vb0 = load_b_frag(ldsV + 0  * NTOK + sstep * 32, NTOK);
    v16bf vb1 = load_b_frag(ldsV + 16 * NTOK + sstep * 32, NTOK);
    o0 = WMMA_BF16(pa, vb0, o0);
    o1 = WMMA_BF16(pa, vb1, o1);
  }

  // store O as bf16 into (B, N, C) for the projection GEMM
#pragma unroll
  for (int r = 0; r < 8; ++r) {
    int token = m0 + r + half * 8;
    size_t rowoff = ((size_t)b * NTOK + token) * DIM + h * HD;
    AO[rowoff + coln]      = (unsigned short)f2bf1(o0[r]);
    AO[rowoff + 16 + coln] = (unsigned short)f2bf1(o1[r]);
  }
}

// ---------------- projection: (65536 x 192) @ (192 x 192) + b -> fp32 out ------
__global__ __launch_bounds__(256) void proj_kernel(
    const unsigned short* __restrict__ AO, const unsigned short* __restrict__ Wpt,
    const float* __restrict__ proj_b, float* __restrict__ out)
{
  int wave = threadIdx.x >> 5;
  int lane = threadIdx.x & 31;
  int half = lane >> 4;
  int coln = lane & 15;
  size_t rowbase = ((size_t)blockIdx.x * 8 + wave) * 16;
  const unsigned short* Arow = AO + rowbase * DIM;

  v16bf afr[6];
#pragma unroll
  for (int ks = 0; ks < 6; ++ks)
    afr[ks] = load_a_frag(Arow + ks * 32, DIM);

  for (int nt = 0; nt < 12; ++nt) {
    __builtin_prefetch(Wpt + (nt + 1) * 16 * DIM, 0, 3);
    v8f acc = {};
#pragma unroll
    for (int ks = 0; ks < 6; ++ks) {
      v16bf bfr = load_b_frag(Wpt + nt * 16 * DIM + ks * 32, DIM);
      acc = WMMA_BF16(afr[ks], bfr, acc);
    }
    int n = nt * 16 + coln;
    float pb = proj_b[n];
#pragma unroll
    for (int r = 0; r < 8; ++r)
      out[(rowbase + r + half * 8) * DIM + n] = acc[r] + pb;
  }
}

// -------------------------------------------------------------------------------
extern "C" void kernel_launch(void* const* d_in, const int* in_sizes, int n_in,
                              void* d_out, int out_size, void* d_ws, size_t ws_size,
                              hipStream_t stream) {
  const float* x          = (const float*)d_in[0];
  const float* mask       = (const float*)d_in[1];
  const float* qkv_w      = (const float*)d_in[2];
  const float* qkv_b      = (const float*)d_in[3];
  const float* proj_w     = (const float*)d_in[4];
  const float* proj_b     = (const float*)d_in[5];
  const float* bias_table = (const float*)d_in[6];
  const int*   rel_index  = (const int*)d_in[7];

  char* ws = (char*)d_ws;
  size_t off = 0;
  auto salloc = [&](size_t bytes) {
    char* p = ws + off;
    off = (off + bytes + 255) & ~(size_t)255;
    return p;
  };
  float*          biasM = (float*)         salloc((size_t)HEADS * NTOK * NTOK * 4);
  unsigned short* Wqkvt = (unsigned short*)salloc((size_t)3 * DIM * DIM * 2);
  unsigned short* Wpt   = (unsigned short*)salloc((size_t)DIM * DIM * 2);
  unsigned short* Q     = (unsigned short*)salloc((size_t)BATCH * HEADS * NTOK * HD * 2);
  unsigned short* K     = (unsigned short*)salloc((size_t)BATCH * HEADS * NTOK * HD * 2);
  unsigned short* Vt    = (unsigned short*)salloc((size_t)BATCH * HEADS * NTOK * HD * 2);
  unsigned short* AO    = (unsigned short*)salloc((size_t)BATCH * NTOK * DIM * 2);

  prep_kernel<<<(HEADS * NTOK * NTOK) / 256, 256, 0, stream>>>(
      qkv_w, proj_w, bias_table, rel_index, biasM, Wqkvt, Wpt);
  qkv_kernel<<<BATCH * 4, 128, 0, stream>>>(x, Wqkvt, qkv_b, Q, K, Vt);
  attn_kernel<<<BATCH * HEADS, 512, 0, stream>>>(Q, K, Vt, biasM, mask, AO);
  proj_kernel<<<(BATCH * NTOK / 16) / 8, 256, 0, stream>>>(AO, Wpt, proj_b, (float*)d_out);
}
